// BiBlock_fp_64682207478090
// MI455X (gfx1250) — matrix operand order repaired and checked by reading, not compiled, not run on hardware
//
#include <hip/hip_runtime.h>
#include <hip/hip_bf16.h>
#include <stdint.h>

// ---------------- problem constants ----------------
#define BB 8
#define NN 4096
#define MM 1024
#define DD 64
#define DOUT 128
#define RBF_THRESH 0.05f

#define DEV static __device__ __forceinline__

typedef __attribute__((ext_vector_type(16))) __bf16 v16bf;
typedef __attribute__((ext_vector_type(8)))  __bf16 v8bf;
typedef __attribute__((ext_vector_type(8)))  float  v8f;
typedef __attribute__((ext_vector_type(4)))  float  v4f;

DEV unsigned lane_id() { return threadIdx.x & 31u; }

DEV v8f vzero8() {
    v8f z;
    #pragma unroll
    for (int i = 0; i < 8; ++i) z[i] = 0.f;
    return z;
}

DEV v8f wmma_bf16(v16bf a, v16bf b, v8f c) {
    // v_wmma_f32_16x16x32_bf16  D = A(16x32) * B(32x16) + C
    return __builtin_amdgcn_wmma_f32_16x16x32_bf16(false, a, false, b, (short)0, c, false, false);
}

// ---- CDNA5 async global->LDS staging (ASYNCcnt-tracked, ISA ch10 / 08_async_tensor) ----
DEV uint32_t lds_off(const void* p) { return (uint32_t)(uintptr_t)p; }  // addr[31:0] is the LDS byte address

DEV void async_b128(uint32_t ldsoff, const void* gptr) {
    asm volatile("global_load_async_to_lds_b128 %0, %1, off"
                 :: "v"(ldsoff), "v"(gptr) : "memory");
}
DEV void wait_async0() { asm volatile("s_wait_asynccnt 0" ::: "memory"); }
DEV void wait_ds0()    { asm volatile("s_wait_dscnt 0" ::: "memory"); }

DEV v16bf cat8(v8bf lo, v8bf hi) {
    return __builtin_shufflevector(lo, hi, 0,1,2,3,4,5,6,7,8,9,10,11,12,13,14,15);
}

// ---- fragment builders (layouts per CDNA5 ISA §7.12.2) ----
// A 16x32 bf16: lane L holds row r=L%16; half=L/16: K = {8h..8h+7} then {16+8h..16+8h+7}
// -> two contiguous 16B reads per lane.
DEV v16bf load_A_bf16_lds(const __bf16* lds, int ld, int row0, int k0) {
    unsigned l = lane_id(); int r = row0 + (l & 15), half = l >> 4;
    const __bf16* p = lds + r * ld + k0 + 8 * half;
    v8bf lo = *(const v8bf*)p;
    v8bf hi = *(const v8bf*)(p + 16);
    return cat8(lo, hi);
}
DEV v16bf load_A_bf16_global(const __bf16* g, int ld, size_t row0, int k0) {
    unsigned l = lane_id(); size_t r = row0 + (l & 15); int half = l >> 4;
    const __bf16* p = g + r * ld + k0 + 8 * half;
    v8bf lo = *(const v8bf*)p;
    v8bf hi = *(const v8bf*)(p + 16);
    return cat8(lo, hi);
}
DEV v16bf load_A_f32_global(const float* g, int ld, size_t row0, int k0) {
    unsigned l = lane_id(); size_t r = row0 + (l & 15); int half = l >> 4;
    const float* p = g + r * ld + k0 + 8 * half;
    v4f f0 = *(const v4f*)p,        f1 = *(const v4f*)(p + 4);
    v4f f2 = *(const v4f*)(p + 16), f3 = *(const v4f*)(p + 20);
    v16bf a;
    #pragma unroll
    for (int i = 0; i < 4; ++i) {
        a[i]      = (__bf16)f0[i];
        a[i + 4]  = (__bf16)f1[i];
        a[i + 8]  = (__bf16)f2[i];
        a[i + 12] = (__bf16)f3[i];
    }
    return a;
}
// B 32x16 bf16 built from an N-major source T[n][k] (i.e. B = T^T):
// lane L holds col n=L%16; K = {8h..8h+7} then {16+8h..16+8h+7} -> two 16B reads per lane.
DEV v16bf load_Bt_bf16_lds(const __bf16* lds, int ld, int col0, int k0) {
    unsigned l = lane_id(); int n = l & 15, half = l >> 4;
    const __bf16* p = lds + (size_t)(col0 + n) * ld + k0 + 8 * half;
    return cat8(*(const v8bf*)p, *(const v8bf*)(p + 16));
}

// ---- half-wave (16-lane) reductions via v_permlane16_b32 (pure VALU, no LDS port) ----
// Selector nibbles: lane i (within its 16-lane row) reads lane i^k. Rows of 16 are
// independent, so lanes 0-15 and 16-31 (different C-matrix rows) reduce separately.
DEV float pl16(float x, unsigned s1, unsigned s2) {
    return __int_as_float(__builtin_amdgcn_permlane16(
        __float_as_int(x), __float_as_int(x), s1, s2, false, false));
}
DEV float hmax16(float x) {
    x = fmaxf(x, pl16(x, 0x67452301u, 0xEFCDAB89u));  // xor 1
    x = fmaxf(x, pl16(x, 0x54761032u, 0xDCFE98BAu));  // xor 2
    x = fmaxf(x, pl16(x, 0x32107654u, 0xBA98FEDCu));  // xor 4
    x = fmaxf(x, pl16(x, 0xFEDCBA98u, 0x76543210u));  // xor 8
    return x;
}
DEV float hsum16(float x) {
    x += pl16(x, 0x67452301u, 0xEFCDAB89u);
    x += pl16(x, 0x54761032u, 0xDCFE98BAu);
    x += pl16(x, 0x32107654u, 0xBA98FEDCu);
    x += pl16(x, 0xFEDCBA98u, 0x76543210u);
    return x;
}

// ================= kernel 1: X(rows x 64) @ W(64x64) -> bf16 out =================
// TMODE=0: out row-major [rows][64].  TMODE=1: out transposed per batch of 1024
// rows: out[(batch*64 + col)*1024 + (r%1024)]  (used for V so P·V B-frags are contiguous).
template <int TMODE>
__global__ __launch_bounds__(256) void proj_gemm_kernel(
    const float* __restrict__ in, const float* __restrict__ W,
    __bf16* __restrict__ out, int rows)
{
    __shared__ __align__(16) __bf16 sWt[64 * 64];   // W^T: sWt[e][d]
    const int tid = threadIdx.x;
    for (int i = tid; i < 64 * 64; i += 256) {
        int e = i >> 6, d = i & 63;
        sWt[i] = (__bf16)W[d * 64 + e];
    }
    __syncthreads();

    const int wave = tid >> 5;
    const size_t row0 = (size_t)blockIdx.x * 128 + wave * 16;
    if (row0 >= (size_t)rows) return;
    const unsigned l = tid & 31; const int nloc = l & 15, half = l >> 4;

    v16bf a0 = load_A_f32_global(in, 64, row0, 0);
    v16bf a1 = load_A_f32_global(in, 64, row0, 32);
    #pragma unroll
    for (int t = 0; t < 4; ++t) {
        v8f acc = vzero8();
        acc = wmma_bf16(a0, load_Bt_bf16_lds(sWt, 64, t * 16, 0),  acc);
        acc = wmma_bf16(a1, load_Bt_bf16_lds(sWt, 64, t * 16, 32), acc);
        const int col = t * 16 + nloc;
        #pragma unroll
        for (int j = 0; j < 8; ++j) {
            const size_t r = row0 + j + 8 * half;
            if (TMODE == 0) {
                out[r * 64 + col] = (__bf16)acc[j];
            } else {
                const size_t batch = r >> 10, mloc = r & 1023;   // rows-per-batch = 1024
                out[(batch * 64 + col) * 1024 + mloc] = (__bf16)acc[j];
            }
        }
    }
}

// ============ kernel 2: fused RBF-gated flash attention -> kern (bf16) ============
// block = 256 thr (8 waves), each wave owns a 16-row q tile; block covers 128 n-rows.
// KV tile = 64 columns/iteration so row-stat reductions & O-rescale amortize 2x.
__global__ __launch_bounds__(256) void attn_kernel(
    const __bf16* __restrict__ qbf, const __bf16* __restrict__ kbf,
    const __bf16* __restrict__ vtbf,   // [b][64][1024] (transposed)
    const float* __restrict__ xyz1, const float* __restrict__ xyz2,
    const float* __restrict__ gamma, __bf16* __restrict__ kern)
{
    __shared__ __align__(16) __bf16 sQ[128 * 64];   // 16 KB
    __shared__ __align__(16) float  sX1[128 * 3];   // 1.5 KB
    __shared__ __align__(16) __bf16 sK[64 * 64];    // 8 KB  row-major [m][d]
    __shared__ __align__(16) __bf16 sVt[64 * 64];   // 8 KB  transposed [d][m]
    __shared__ __align__(16) float  sX2[64 * 3];    // 768 B
    __shared__ __align__(16) __bf16 sP[8][16 * 64]; // 16 KB (per-wave P staging)

    const int tid = threadIdx.x, wave = tid >> 5;
    const unsigned l = tid & 31; const int nloc = l & 15, half = l >> 4;
    const int batch = blockIdx.y;
    const int row0  = blockIdx.x * 128;     // first n-row of this block
    const float g   = gamma[0];

    // --- stage q tile + xyz1 tile via async DMA ---
    {
        const char* gq = (const char*)(qbf + ((size_t)batch * NN + row0) * DD);
        for (int i = tid; i < (128 * 64 * 2) / 16; i += 256)
            async_b128(lds_off(sQ) + i * 16, gq + i * 16);
        const char* gx = (const char*)(xyz1 + ((size_t)batch * NN + row0) * 3);
        for (int i = tid; i < (128 * 3 * 4) / 16; i += 256)
            async_b128(lds_off(sX1) + i * 16, gx + i * 16);
    }
    wait_async0();
    __syncthreads();

    v16bf qa0 = load_A_bf16_lds(sQ, 64, wave * 16, 0);
    v16bf qa1 = load_A_bf16_lds(sQ, 64, wave * 16, 32);

    v8f O0 = vzero8(), O1 = vzero8(), O2 = vzero8(), O3 = vzero8();
    float mrow[8], lrow[8];
    #pragma unroll
    for (int j = 0; j < 8; ++j) { mrow[j] = -3.0e38f; lrow[j] = 0.f; }

    for (int m0 = 0; m0 < MM; m0 += 64) {
        __syncthreads();   // all waves done reading previous sK/sVt
        const char* gk  = (const char*)(kbf  + ((size_t)batch * MM + m0) * DD);
        const char* gx2 = (const char*)(xyz2 + ((size_t)batch * MM + m0) * 3);
        // K tile: 64 rows x 64 d, contiguous 8KB
        for (int i = tid; i < (64 * 64 * 2) / 16; i += 256)
            async_b128(lds_off(sK) + i * 16, gk + i * 16);
        // V^T tile: 64 d-rows x 64 m, each global row strided by 1024 elems
        {
            const char* gv = (const char*)(vtbf + ((size_t)batch * 64) * 1024 + m0);
            for (int i = tid; i < 512; i += 256) {     // 64 rows * 8 chunks of 16B
                int rowd = i >> 3, part = i & 7;
                async_b128(lds_off(sVt) + rowd * 128 + part * 16,
                           gv + (size_t)rowd * (1024 * 2) + part * 16);
            }
        }
        for (int i = tid; i < (64 * 3 * 4) / 16; i += 256)
            async_b128(lds_off(sX2) + i * 16, gx2 + i * 16);
        wait_async0();
        __syncthreads();

        // S = q @ k^T  (four 16-col subtiles of the 64-wide m chunk)
        v8f S0 = vzero8(), S1 = vzero8(), S2 = vzero8(), S3 = vzero8();
        S0 = wmma_bf16(qa0, load_Bt_bf16_lds(sK, 64, 0,  0),  S0);
        S0 = wmma_bf16(qa1, load_Bt_bf16_lds(sK, 64, 0,  32), S0);
        S1 = wmma_bf16(qa0, load_Bt_bf16_lds(sK, 64, 16, 0),  S1);
        S1 = wmma_bf16(qa1, load_Bt_bf16_lds(sK, 64, 16, 32), S1);
        S2 = wmma_bf16(qa0, load_Bt_bf16_lds(sK, 64, 32, 0),  S2);
        S2 = wmma_bf16(qa1, load_Bt_bf16_lds(sK, 64, 32, 32), S2);
        S3 = wmma_bf16(qa0, load_Bt_bf16_lds(sK, 64, 48, 0),  S3);
        S3 = wmma_bf16(qa1, load_Bt_bf16_lds(sK, 64, 48, 32), S3);

        const int c0 = nloc, c1 = 16 + nloc, c2 = 32 + nloc, c3 = 48 + nloc;
        const float x20 = sX2[c0 * 3], y20 = sX2[c0 * 3 + 1], z20 = sX2[c0 * 3 + 2];
        const float x21 = sX2[c1 * 3], y21 = sX2[c1 * 3 + 1], z21 = sX2[c1 * 3 + 2];
        const float x22 = sX2[c2 * 3], y22 = sX2[c2 * 3 + 1], z22 = sX2[c2 * 3 + 2];
        const float x23 = sX2[c3 * 3], y23 = sX2[c3 * 3 + 1], z23 = sX2[c3 * 3 + 2];

        #pragma unroll
        for (int j = 0; j < 8; ++j) {
            const int rl = wave * 16 + j + 8 * half;      // row within block
            float s0 = S0[j], s1 = S1[j], s2 = S2[j], s3 = S3[j];
            // online softmax over att2 only (denominator excludes att1)
            float tmax  = hmax16(fmaxf(fmaxf(s0, s1), fmaxf(s2, s3)));
            float mnew  = fmaxf(mrow[j], tmax);
            float scale = __expf(mrow[j] - mnew);
            mrow[j] = mnew;
            float p0 = __expf(s0 - mnew), p1 = __expf(s1 - mnew);
            float p2 = __expf(s2 - mnew), p3 = __expf(s3 - mnew);
            lrow[j] = lrow[j] * scale + hsum16((p0 + p1) + (p2 + p3));
            O0[j] *= scale; O1[j] *= scale; O2[j] *= scale; O3[j] *= scale;
            // RBF gate from positions
            const float x1 = sX1[rl * 3], y1 = sX1[rl * 3 + 1], z1 = sX1[rl * 3 + 2];
            float dx = x1 - x20, dy = y1 - y20, dz = z1 - z20;
            float e0 = __expf(-g * (dx * dx + dy * dy + dz * dz));
            e0 = (e0 <= RBF_THRESH) ? 0.f : e0;
            dx = x1 - x21; dy = y1 - y21; dz = z1 - z21;
            float e1 = __expf(-g * (dx * dx + dy * dy + dz * dz));
            e1 = (e1 <= RBF_THRESH) ? 0.f : e1;
            dx = x1 - x22; dy = y1 - y22; dz = z1 - z22;
            float e2 = __expf(-g * (dx * dx + dy * dy + dz * dz));
            e2 = (e2 <= RBF_THRESH) ? 0.f : e2;
            dx = x1 - x23; dy = y1 - y23; dz = z1 - z23;
            float e3 = __expf(-g * (dx * dx + dy * dy + dz * dz));
            e3 = (e3 <= RBF_THRESH) ? 0.f : e3;
            // stage gated P (16x64 bf16, row-major) for the A operand of P·V
            const int prow = j + 8 * half;
            sP[wave][prow * 64 + c0] = (__bf16)(e0 * p0);
            sP[wave][prow * 64 + c1] = (__bf16)(e1 * p1);
            sP[wave][prow * 64 + c2] = (__bf16)(e2 * p2);
            sP[wave][prow * 64 + c3] = (__bf16)(e3 * p3);
        }
        wait_ds0();  // wave-local LDS RAW (sP is private to this wave)

        v16bf pa0 = load_A_bf16_lds(sP[wave], 64, 0, 0);   // P cols 0..31
        v16bf pa1 = load_A_bf16_lds(sP[wave], 64, 0, 32);  // P cols 32..63
        O0 = wmma_bf16(pa0, load_Bt_bf16_lds(sVt, 64, 0,  0),  O0);
        O0 = wmma_bf16(pa1, load_Bt_bf16_lds(sVt, 64, 0,  32), O0);
        O1 = wmma_bf16(pa0, load_Bt_bf16_lds(sVt, 64, 16, 0),  O1);
        O1 = wmma_bf16(pa1, load_Bt_bf16_lds(sVt, 64, 16, 32), O1);
        O2 = wmma_bf16(pa0, load_Bt_bf16_lds(sVt, 64, 32, 0),  O2);
        O2 = wmma_bf16(pa1, load_Bt_bf16_lds(sVt, 64, 32, 32), O2);
        O3 = wmma_bf16(pa0, load_Bt_bf16_lds(sVt, 64, 48, 0),  O3);
        O3 = wmma_bf16(pa1, load_Bt_bf16_lds(sVt, 64, 48, 32), O3);
    }

    // kern = O / l
    #pragma unroll
    for (int j = 0; j < 8; ++j) {
        const float inv = 1.f / lrow[j];
        const size_t r = (size_t)batch * NN + row0 + wave * 16 + j + 8 * half;
        kern[r * 64 +  0 + nloc] = (__bf16)(O0[j] * inv);
        kern[r * 64 + 16 + nloc] = (__bf16)(O1[j] * inv);
        kern[r * 64 + 32 + nloc] = (__bf16)(O2[j] * inv);
        kern[r * 64 + 48 + nloc] = (__bf16)(O3[j] * inv);
    }
}

// ========== kernel 3: out = relu([points1 | kern] @ Wo + bo), f32 out ==========
__global__ __launch_bounds__(256) void out_mlp_kernel(
    const float* __restrict__ points1, const __bf16* __restrict__ kern,
    const float* __restrict__ Wo, const float* __restrict__ bo,
    float* __restrict__ out)
{
    __shared__ __align__(16) __bf16 sWt[128 * 128];  // Wo^T: sWt[o][c], 32 KB
    const int tid = threadIdx.x;
    for (int i = tid; i < 128 * 128; i += 256) {
        int o = i >> 7, c = i & 127;
        sWt[i] = (__bf16)Wo[c * 128 + o];
    }
    __syncthreads();

    const int wave = tid >> 5;
    const unsigned l = tid & 31; const int nloc = l & 15, half = l >> 4;
    const size_t row0 = (size_t)blockIdx.x * 128 + wave * 16;  // over b*n rows

    v16bf a0 = load_A_f32_global (points1, 64, row0, 0);
    v16bf a1 = load_A_f32_global (points1, 64, row0, 32);
    v16bf a2 = load_A_bf16_global(kern,    64, row0, 0);
    v16bf a3 = load_A_bf16_global(kern,    64, row0, 32);

    #pragma unroll
    for (int t = 0; t < 8; ++t) {
        v8f acc = vzero8();
        acc = wmma_bf16(a0, load_Bt_bf16_lds(sWt, 128, t * 16, 0),  acc);
        acc = wmma_bf16(a1, load_Bt_bf16_lds(sWt, 128, t * 16, 32), acc);
        acc = wmma_bf16(a2, load_Bt_bf16_lds(sWt, 128, t * 16, 64), acc);
        acc = wmma_bf16(a3, load_Bt_bf16_lds(sWt, 128, t * 16, 96), acc);
        const int col = t * 16 + nloc;
        const float bb = bo[col];
        #pragma unroll
        for (int j = 0; j < 8; ++j) {
            float v = acc[j] + bb;
            out[(row0 + j + 8 * half) * DOUT + col] = v > 0.f ? v : 0.f;
        }
    }
}

// =============================== launcher ===============================
extern "C" void kernel_launch(void* const* d_in, const int* in_sizes, int n_in,
                              void* d_out, int out_size, void* d_ws, size_t ws_size,
                              hipStream_t stream) {
    const float* xyz1    = (const float*)d_in[0];
    const float* xyz2    = (const float*)d_in[1];
    const float* points1 = (const float*)d_in[2];
    const float* points2 = (const float*)d_in[3];
    const float* Wq      = (const float*)d_in[4];
    const float* Wk      = (const float*)d_in[5];
    const float* Wv      = (const float*)d_in[6];
    const float* gamma   = (const float*)d_in[7];
    const float* Wo      = (const float*)d_in[8];
    const float* bo      = (const float*)d_in[9];

    // workspace layout (bf16): q | k | v^T | kern  (~10.5 MB total)
    __bf16* qbf  = (__bf16*)d_ws;
    __bf16* kbf  = qbf + (size_t)BB * NN * DD;
    __bf16* vtbf = kbf + (size_t)BB * MM * DD;      // stored [b][64][1024]
    __bf16* kern = vtbf + (size_t)BB * MM * DD;

    proj_gemm_kernel<0><<<(BB * NN) / 128, 256, 0, stream>>>(points1, Wq, qbf, BB * NN);
    proj_gemm_kernel<0><<<(BB * MM) / 128, 256, 0, stream>>>(points2, Wk, kbf, BB * MM);
    proj_gemm_kernel<1><<<(BB * MM) / 128, 256, 0, stream>>>(points2, Wv, vtbf, BB * MM);

    attn_kernel<<<dim3(NN / 128, BB), 256, 0, stream>>>(qbf, kbf, vtbf, xyz1, xyz2, gamma, kern);

    out_mlp_kernel<<<(BB * NN) / 128, 256, 0, stream>>>(points1, kern, Wo, bo, (float*)d_out);
}
